// MessageAggregator_76879914598571
// MI455X (gfx1250) — matrix operand order repaired and checked by reading, not compiled, not run on hardware
//
#include <hip/hip_runtime.h>
#include <math.h>

#define D 512
#define NMSG 32
#define KSTEP 32
#define TILE 16
#define NKB (D / KSTEP)   /* 16 K-blocks  */
#define NNB (D / TILE)    /* 32 N-blocks  */
#define SLAB_HALVES (NNB * 32 * 16)      /* 16384 f16 = 32KB per kBlock slab */
#define INV_SCALE 0.044194173824159216f  /* 1/sqrt(512) */

typedef _Float16 v16h __attribute__((ext_vector_type(16)));
typedef float    v8f  __attribute__((ext_vector_type(8)));
typedef unsigned int tdm_v4u __attribute__((ext_vector_type(4)));
typedef int          tdm_v8i __attribute__((ext_vector_type(8)));
typedef int          tdm_v4i __attribute__((ext_vector_type(4)));

#if defined(__has_builtin)
#if __has_builtin(__builtin_amdgcn_tensor_load_to_lds) && __has_builtin(__builtin_amdgcn_s_wait_tensorcnt)
#define HAVE_TDM 1
#endif
#endif
#ifndef HAVE_TDM
#define HAVE_TDM 0
#endif

#define WMMA(a, b, c) __builtin_amdgcn_wmma_f32_16x16x32_f16(false, (a), false, (b), (short)0, (c), false, false)

// ---------------- packed-fragment addressing (32B per lane per fragment) ----------------
__device__ __forceinline__ size_t aoff(int blockRow, int kBlock, int lane) {
  return ((size_t)(blockRow * NKB + kBlock) * 32 + lane) * 16;
}
__device__ __forceinline__ size_t boff(int kBlock, int nBlock, int lane) {
  return ((size_t)(kBlock * NNB + nBlock) * 32 + lane) * 16;
}

// ---------------- f32 -> WMMA fragment gathers (one-time prep/pack only) ----------------
__device__ __forceinline__ v16h frag_a(const float* __restrict__ P, int ld, int row0, int k0, int lane) {
  const int m  = lane & 15;
  const int kb = (lane & 16) ? 8 : 0;
  v16h f;
#pragma unroll
  for (int j = 0; j < 8; ++j) {
    const int kk = ((j < 4) ? (2 * j) : (16 + 2 * (j - 4))) + kb;
    const float* p = P + (size_t)(row0 + m) * ld + (k0 + kk);
    f[2 * j]     = (_Float16)p[0];
    f[2 * j + 1] = (_Float16)p[1];
  }
  return f;
}
__device__ __forceinline__ v16h frag_a_T(const float* __restrict__ P, int ld, int row0, int k0, int lane) {
  const int m  = lane & 15;
  const int kb = (lane & 16) ? 8 : 0;
  v16h f;
#pragma unroll
  for (int j = 0; j < 8; ++j) {
    const int kk = ((j < 4) ? (2 * j) : (16 + 2 * (j - 4))) + kb;
    const float* p = P + (size_t)(k0 + kk) * ld + (row0 + m);
    f[2 * j]     = (_Float16)p[0];
    f[2 * j + 1] = (_Float16)p[ld];
  }
  return f;
}
template <bool TRANS>
__device__ __forceinline__ v16h frag_b(const float* __restrict__ P, int ld, int k0, int n0, int lane) {
  const int n  = lane & 15;
  const int kb = (lane & 16) ? 8 : 0;
  v16h f;
#pragma unroll
  for (int j = 0; j < 8; ++j) {
    const int k = k0 + ((j < 4) ? (2 * j) : (16 + 2 * (j - 4))) + kb;
    if (TRANS) {
      const float* p = P + (size_t)(n0 + n) * ld + k;
      f[2 * j] = (_Float16)p[0];  f[2 * j + 1] = (_Float16)p[1];
    } else {
      const float* p = P + (size_t)k * ld + (n0 + n);
      f[2 * j] = (_Float16)p[0];  f[2 * j + 1] = (_Float16)p[ld];
    }
  }
  return f;
}
__device__ __forceinline__ void store_tile(float* __restrict__ P, int ld, int row0, int n0, int lane, v8f c) {
  const int n  = lane & 15;
  const int rb = (lane & 16) ? 8 : 0;
#pragma unroll
  for (int r = 0; r < 8; ++r) P[(size_t)(row0 + rb + r) * ld + n0 + n] = c[r];
}

// ---------------- TDM: DMA one contiguous 32KB slab (16384 f16) global -> LDS ----------------
#if HAVE_TDM
__device__ __forceinline__ void tdm_load_slab(const _Float16* __restrict__ gsrc, unsigned int ldsOff) {
  const unsigned long long ga = (unsigned long long)gsrc;
  tdm_v4u g0 = {};
  g0[0] = 1u;                                            // count=1 (valid descriptor)
  g0[1] = ldsOff;                                        // lds_addr
  g0[2] = (unsigned int)ga;                              // global_addr[31:0]
  g0[3] = (unsigned int)((ga >> 32) & 0x01FFFFFFu) | (2u << 30);  // addr[56:32] | type=2
  tdm_v8i g1 = {};
  g1[0] = (1 << 16);                                     // data_size=1 (2 bytes), wg_mask=0
  g1[1] = (0x4000 << 16);                                // tensor_dim0 = 16384 (low16 @ bit48)
  g1[2] = (1 << 16);                                     // tensor_dim0 hi=0 | tensor_dim1 = 1
  g1[3] = (0x4000 << 16);                                // tensor_dim1 hi=0 | tile_dim0 = 16384
  g1[4] = 0;                                             // tile_dim1 = 0, tile_dim2 = 0 (unused)
  g1[5] = 16384;                                         // tensor_dim0_stride low32
  g1[6] = 0;                                             // stride hi | tensor_dim1_stride low
  g1[7] = 0;
  tdm_v4i z4 = {};
#if __clang_major__ >= 23
  tdm_v8i z8 = {};
  __builtin_amdgcn_tensor_load_to_lds(g0, g1, z4, z4, z8, 0);
#else
  __builtin_amdgcn_tensor_load_to_lds(g0, g1, z4, z4, 0);
#endif
}
#endif

// ---------------- K1: A = Wq^T Wk ; Bm = Wo Wv (one 16x16 tile per wave, uniform split) ----------------
template <bool TA>
__device__ __forceinline__ void prep_tile(const float* __restrict__ WA, const float* __restrict__ WB,
                                          float* __restrict__ Outm, int idx, int lane) {
  const int row0 = (idx >> 5) << 4;
  const int n0   = (idx & 31) << 4;
  v8f acc = {};
  for (int k0 = 0; k0 < D; k0 += KSTEP) {
    v16h a = TA ? frag_a_T(WA, D, row0, k0, lane) : frag_a(WA, D, row0, k0, lane);
    v16h b = frag_b<false>(WB, D, k0, n0, lane);
    acc = WMMA(a, b, acc);
  }
  store_tile(Outm, D, row0, n0, lane, acc);
}

__global__ void __launch_bounds__(256) mag_prep(const float* __restrict__ Wq, const float* __restrict__ Wk,
                                                const float* __restrict__ Wo, const float* __restrict__ Wv,
                                                float* __restrict__ A, float* __restrict__ Bm) {
  const int lane = threadIdx.x & 31;
  const int w    = threadIdx.x >> 5;
  if (blockIdx.x < 128) {
    prep_tile<true >(Wq, Wk, A,  blockIdx.x * 8 + w, lane);
  } else {
    prep_tile<false>(Wo, Wv, Bm, (blockIdx.x - 128) * 8 + w, lane);
  }
}

// ---------------- pack kernels: f32 row-major -> packed f16 fragments ----------------
__global__ void __launch_bounds__(256) mag_pack_a(const float* __restrict__ src, _Float16* __restrict__ dst,
                                                  int rows) {
  const int wid  = blockIdx.x * 8 + (threadIdx.x >> 5);
  const int lane = threadIdx.x & 31;
  const int blockRow = wid / NKB, kBlock = wid % NKB;
  if (blockRow * 16 >= rows) return;
  v16h f = frag_a(src, D, blockRow * 16, kBlock * KSTEP, lane);
  *(v16h*)(dst + aoff(blockRow, kBlock, lane)) = f;
}

template <bool TRANS>
__global__ void __launch_bounds__(256) mag_pack_b(const float* __restrict__ src, _Float16* __restrict__ dst) {
  const int wid  = blockIdx.x * 8 + (threadIdx.x >> 5);
  const int lane = threadIdx.x & 31;
  const int kBlock = wid / NNB, nBlock = wid % NNB;
  v16h f = frag_b<TRANS>(src, D, kBlock * KSTEP, nBlock * TILE, lane);
  *(v16h*)(dst + boff(kBlock, nBlock, lane)) = f;
}

// ---------------- K2: t = h @ A  (TDM double-buffers the B slab into LDS; wmma from ds_load) ----------------
__global__ void __launch_bounds__(256) mag_tproj(const _Float16* __restrict__ pH, const _Float16* __restrict__ pA,
                                                 float* __restrict__ T) {
  const int lane = threadIdx.x & 31;
  const int w    = threadIdx.x >> 5;
  const int blockRow = blockIdx.x;
  v8f acc[4] = {};
#if HAVE_TDM
  __shared__ _Float16 sB[2][SLAB_HALVES];   // 2 x 32KB double buffer
  if (threadIdx.x < 32)
    tdm_load_slab(pA, (unsigned int)(unsigned long long)(void*)&sB[0][0]);
  for (int kB = 0; kB < NKB; ++kB) {
    if (threadIdx.x < 32) __builtin_amdgcn_s_wait_tensorcnt(0);
    __syncthreads();                        // slab kB resident for whole WG
    if (threadIdx.x < 32 && kB + 1 < NKB)
      tdm_load_slab(pA + (size_t)(kB + 1) * SLAB_HALVES,
                    (unsigned int)(unsigned long long)(void*)&sB[(kB + 1) & 1][0]);
    const v16h a = *(const v16h*)(pH + aoff(blockRow, kB, lane));
    const _Float16* slab = &sB[kB & 1][0];
#pragma unroll
    for (int t = 0; t < 4; ++t) {
      const v16h b = *(const v16h*)(slab + (((w * 4 + t) * 32 + lane) * 16));
      acc[t] = WMMA(a, b, acc[t]);
    }
    __syncthreads();                        // everyone done before kB+2 overwrites this buffer
  }
#else
  for (int kB = 0; kB < NKB; ++kB) {
    const v16h a = *(const v16h*)(pH + aoff(blockRow, kB, lane));
#pragma unroll
    for (int t = 0; t < 4; ++t) {
      const v16h b = *(const v16h*)(pA + boff(kB, w * 4 + t, lane));
      acc[t] = WMMA(a, b, acc[t]);
    }
  }
#endif
#pragma unroll
  for (int t = 0; t < 4; ++t) store_tile(T, D, blockRow * 16, (w * 4 + t) * 16, lane, acc[t]);
}

// ---------------- K3: single-pass online-softmax aggregation (1 wave = 1 agent) ----------------
__global__ void __launch_bounds__(256) mag_attn(const float* __restrict__ T, const float* __restrict__ M,
                                                float* __restrict__ Mbar, int Btot) {
  const int lane = threadIdx.x & 31;
  const int b    = blockIdx.x * 8 + (threadIdx.x >> 5);
  if (b >= Btot) return;
  float4 tq[4], acc[4];
  const float4* trow = (const float4*)(T + (size_t)b * D) + lane * 4;
#pragma unroll
  for (int q = 0; q < 4; ++q) { tq[q] = trow[q]; acc[q] = make_float4(0.f, 0.f, 0.f, 0.f); }
  float runMax = -3.4e38f, Z = 0.f;
  const float* mb = M + (size_t)b * NMSG * D;
  for (int n = 0; n < NMSG; ++n) {
    const float4* mrow = (const float4*)(mb + (size_t)n * D) + lane * 4;
    if (n + 1 < NMSG) __builtin_prefetch((const void*)(mb + (size_t)(n + 1) * D + lane * 16), 0, 1);
    float4 mv[4]; float part = 0.f;
#pragma unroll
    for (int q = 0; q < 4; ++q) {
      mv[q] = mrow[q];
      part = fmaf(mv[q].x, tq[q].x, part); part = fmaf(mv[q].y, tq[q].y, part);
      part = fmaf(mv[q].z, tq[q].z, part); part = fmaf(mv[q].w, tq[q].w, part);
    }
#pragma unroll
    for (int off = 16; off >= 1; off >>= 1) part += __shfl_xor(part, off, 32);
    const float s = part * INV_SCALE;   // wave-uniform -> uniform branch below
    if (s > runMax) {                   // new max: rescale history (also taken on first iter)
      const float corr = __expf(runMax - s);   // 0 on first iter
      Z = Z * corr + 1.f;
#pragma unroll
      for (int q = 0; q < 4; ++q) {
        acc[q].x = fmaf(acc[q].x, corr, mv[q].x);
        acc[q].y = fmaf(acc[q].y, corr, mv[q].y);
        acc[q].z = fmaf(acc[q].z, corr, mv[q].z);
        acc[q].w = fmaf(acc[q].w, corr, mv[q].w);
      }
      runMax = s;
    } else {                            // common case: no rescale
      const float wgt = __expf(s - runMax);
      Z += wgt;
#pragma unroll
      for (int q = 0; q < 4; ++q) {
        acc[q].x = fmaf(wgt, mv[q].x, acc[q].x);
        acc[q].y = fmaf(wgt, mv[q].y, acc[q].y);
        acc[q].z = fmaf(wgt, mv[q].z, acc[q].z);
        acc[q].w = fmaf(wgt, mv[q].w, acc[q].w);
      }
    }
  }
  const float inv = 1.f / Z;
  float4* orow = (float4*)(Mbar + (size_t)b * D) + lane * 4;
#pragma unroll
  for (int q = 0; q < 4; ++q)
    orow[q] = make_float4(acc[q].x * inv, acc[q].y * inv, acc[q].z * inv, acc[q].w * inv);
}

// ---------------- K4: agg = mbar@Bm^T + bo -> LayerNorm; gate = sigmoid(h@Wg^T+bg); out = gate*LN ----------------
__global__ void __launch_bounds__(256) mag_out(const _Float16* __restrict__ pH,  const _Float16* __restrict__ pM,
                                               const _Float16* __restrict__ pBmT, const _Float16* __restrict__ pWgT,
                                               const float* __restrict__ bo, const float* __restrict__ bg,
                                               const float* __restrict__ gamma, const float* __restrict__ beta,
                                               float* __restrict__ Out) {
  __shared__ float s_agg[16 * D];
  __shared__ float s_p1[16 * 16];
  __shared__ float s_p2[16 * 16];
  __shared__ float s_mu[16];
  __shared__ float s_rs[16];
  const int tid  = threadIdx.x;
  const int lane = tid & 31;
  const int w    = tid >> 5;
  const int blockRow = blockIdx.x;
  const int row0 = blockRow * TILE;
  v8f acc[4] = {}, gacc[4] = {};
  for (int kB = 0; kB < NKB; ++kB) {
    const v16h am = *(const v16h*)(pM + aoff(blockRow, kB, lane));
    const v16h ah = *(const v16h*)(pH + aoff(blockRow, kB, lane));
#pragma unroll
    for (int t = 0; t < 4; ++t) {
      const v16h b1 = *(const v16h*)(pBmT + boff(kB, w * 4 + t, lane));
      acc[t]  = WMMA(am, b1, acc[t]);
      const v16h b2 = *(const v16h*)(pWgT + boff(kB, w * 4 + t, lane));
      gacc[t] = WMMA(ah, b2, gacc[t]);
    }
  }
  const int rb = (lane & 16) ? 8 : 0;
  const int nc = lane & 15;
#pragma unroll
  for (int t = 0; t < 4; ++t) {
    const int col = (w * 4 + t) * 16 + nc;
    const float bias = bo[col];
#pragma unroll
    for (int r = 0; r < 8; ++r) s_agg[(rb + r) * D + col] = acc[t][r] + bias;
  }
  __syncthreads();
  const int row = tid >> 4, j = tid & 15;
  float s1 = 0.f, s2 = 0.f;
  for (int c = j; c < D; c += 16) { const float v = s_agg[row * D + c]; s1 += v; s2 += v * v; }
  s_p1[row * 16 + j] = s1; s_p2[row * 16 + j] = s2;
  __syncthreads();
  if (j == 0) {
    float a = 0.f, q = 0.f;
    for (int u = 0; u < 16; ++u) { a += s_p1[row * 16 + u]; q += s_p2[row * 16 + u]; }
    const float mu  = a * (1.f / (float)D);
    const float var = q * (1.f / (float)D) - mu * mu;
    s_mu[row] = mu; s_rs[row] = rsqrtf(var + 1e-5f);
  }
  __syncthreads();
  for (int c = j; c < D; c += 16) {
    const int i = row * D + c;
    s_agg[i] = (s_agg[i] - s_mu[row]) * s_rs[row] * gamma[c] + beta[c];
  }
  __syncthreads();
#pragma unroll
  for (int t = 0; t < 4; ++t) {
    const int col = (w * 4 + t) * 16 + nc;
    const float gb = bg[col];
#pragma unroll
    for (int r = 0; r < 8; ++r) {
      const int rr = rb + r;
      const float g = 1.f / (1.f + __expf(-(gacc[t][r] + gb)));
      Out[(size_t)(row0 + rr) * D + col] = g * s_agg[rr * D + col];
    }
  }
}

extern "C" void kernel_launch(void* const* d_in, const int* in_sizes, int n_in,
                              void* d_out, int out_size, void* d_ws, size_t ws_size,
                              hipStream_t stream) {
  const float* H     = (const float*)d_in[0];
  const float* Msg   = (const float*)d_in[1];
  const float* Wq    = (const float*)d_in[2];
  const float* Wk    = (const float*)d_in[3];
  const float* Wv    = (const float*)d_in[4];
  const float* Wo    = (const float*)d_in[5];
  const float* bo    = (const float*)d_in[6];
  const float* gamma = (const float*)d_in[7];
  const float* beta  = (const float*)d_in[8];
  const float* Wg    = (const float*)d_in[9];
  const float* bg    = (const float*)d_in[10];
  float* Out = (float*)d_out;
  const int Btot = in_sizes[0] / D;   // 4096

  float* A    = (float*)d_ws;
  float* Bm   = A  + (size_t)D * D;
  float* T    = Bm + (size_t)D * D;
  float* Mbar = T  + (size_t)Btot * D;
  _Float16* pH    = (_Float16*)(Mbar + (size_t)Btot * D);
  _Float16* pMbar = pH    + (size_t)Btot * D;
  _Float16* pA    = pMbar + (size_t)Btot * D;
  _Float16* pBmT  = pA    + (size_t)D * D;
  _Float16* pWgT  = pBmT  + (size_t)D * D;

  const int packAblocks = (Btot / 16) * NKB / 8;

  mag_prep         <<<dim3(256),          dim3(256), 0, stream>>>(Wq, Wk, Wo, Wv, A, Bm);
  mag_pack_a       <<<dim3(packAblocks),  dim3(256), 0, stream>>>(H, pH, Btot);
  mag_pack_b<false><<<dim3(64),           dim3(256), 0, stream>>>(A,  pA);
  mag_pack_b<true> <<<dim3(64),           dim3(256), 0, stream>>>(Bm, pBmT);
  mag_pack_b<true> <<<dim3(64),           dim3(256), 0, stream>>>(Wg, pWgT);
  mag_tproj        <<<dim3(Btot / 16),    dim3(256), 0, stream>>>(pH, pA, T);
  mag_attn         <<<dim3((Btot + 7)/8), dim3(256), 0, stream>>>(T, Msg, Mbar, Btot);
  mag_pack_a       <<<dim3(packAblocks),  dim3(256), 0, stream>>>(Mbar, pMbar, Btot);
  mag_out          <<<dim3(Btot / 16),    dim3(256), 0, stream>>>(pH, pMbar, pBmT, pWgT, bo, bg, gamma, beta, Out);
}
